// ResnetBlocWithAttn_9474697855255
// MI455X (gfx1250) — compile-verified, matrix operand order checked by reading, not run on hardware
//
#include <hip/hip_runtime.h>
#include <hip/hip_bf16.h>
#include <math.h>

// ---------------------------------------------------------------------------
// CDNA5 (gfx1250) implementation. All matmul work via v_wmma_f32_16x16x32_bf16.
// bf16 activations are stored PIXEL-MAJOR [B][HW][C] so GEMM operand gathers
// are contiguous 16B vector loads. GEMM tiles are staged with double-buffered
// GLOBAL_LOAD_ASYNC_TO_LDS_B128 copies (ASYNCcnt), overlapping DMA with WMMA.
// ---------------------------------------------------------------------------

typedef __bf16 bf16_t;
typedef __attribute__((ext_vector_type(16))) __bf16 v16bf;
typedef __attribute__((ext_vector_type(8)))  __bf16 bf16x8;
typedef __attribute__((ext_vector_type(8)))  float  v8f;

#define KP 40    // LDS K-pitch for 32-wide K tiles (80B rows, 16B aligned)
#define QP 264   // LDS K-pitch for 256-wide K tiles (528B rows, 16B aligned)
#define VP 72    // LDS K-pitch for 64-wide K tiles (144B rows, 16B aligned)

// Async copy of 16 bytes per lane, global -> LDS (gfx1250 async path).
// lds_off = LDS byte offset (low 32 bits of generic shared address, ISA 10.2).
__device__ __forceinline__ void async_b128(unsigned lds_off, const void* gsrc) {
  asm volatile("global_load_async_to_lds_b128 %0, %1, off"
               :: "v"(lds_off), "v"(gsrc) : "memory");
}
// Async loads complete in order -> waiting ASYNCcnt<=4 guarantees the
// previously-issued tile (4 copies) has landed while the next 4 fly.
__device__ __forceinline__ void wait_async4() {
  asm volatile("s_wait_asynccnt 0x4" ::: "memory");
}
__device__ __forceinline__ void wait_async0() {
  asm volatile("s_wait_asynccnt 0x0" ::: "memory");
}
__device__ __forceinline__ unsigned lds_addr(const void* p) {
  return (unsigned)(unsigned long long)p;
}

// Build one 16x32 bf16 WMMA operand fragment from an LDS row (K contiguous).
// Per ISA 7.12.2: lanes 0-15 hold K {0..7,16..23}, lanes 16-31 hold {8..15,24..31}.
__device__ __forceinline__ v16bf load_frag(const bf16_t* rowp, int lh) {
  const int b1 = lh * 8;
  const int b2 = 16 + lh * 8;
  bf16x8 lo = *(const bf16x8*)(rowp + b1);
  bf16x8 hi = *(const bf16x8*)(rowp + b2);
  v16bf r;
#pragma unroll
  for (int i = 0; i < 8; i++) { r[i] = lo[i]; r[i + 8] = hi[i]; }
  return r;
}

__device__ __forceinline__ v8f wmma_bf16(v16bf a, v16bf b, v8f c) {
  return __builtin_amdgcn_wmma_f32_16x16x32_bf16(false, a, false, b, (short)0, c,
                                                 false, false);
}

// ---------------------------------------------------------------------------
// Weight repack kernels (fp32 -> bf16, GEMM-friendly layouts)
// ---------------------------------------------------------------------------
// conv w [Co=256][Ci=256][3][3] -> [p=9][Co][Ci] bf16
__global__ void pack3x3_kernel(const float* __restrict__ w, bf16_t* __restrict__ out) {
  int o = blockIdx.x * blockDim.x + threadIdx.x;      // 9*256*256
  int p = o >> 16;
  int rem = o & 65535;
  int co = rem >> 8, ci = rem & 255;
  out[o] = (bf16_t)w[(size_t)(co * 256 + ci) * 9 + p];
}

__global__ void pack_cast_kernel(const float* __restrict__ w, bf16_t* __restrict__ out,
                                 int n) {
  int o = blockIdx.x * blockDim.x + threadIdx.x;
  if (o < n) out[o] = (bf16_t)w[o];
}

__global__ void zero_pad_kernel(float* __restrict__ p) {
  p[threadIdx.x] = 0.f;      // 64 threads -> 256B of zeros (OOB redirect target)
}

// ---------------------------------------------------------------------------
// FiLM: film[b][co] = noise_embedding[b] @ noise_w + noise_b   (16x256, tiny)
// ---------------------------------------------------------------------------
__global__ void film_kernel(const float* __restrict__ ne, const float* __restrict__ nw,
                            const float* __restrict__ nb, float* __restrict__ film) {
  int b = blockIdx.x;
  int co = threadIdx.x;
  float acc = nb[co];
  for (int e = 0; e < 256; e++) acc += ne[b * 256 + e] * nw[e * 256 + co];
  film[b * 256 + co] = acc;
}

// ---------------------------------------------------------------------------
// GroupNorm (32 groups) [+FiLM pre-add] [+SiLU].
// fp32 channel-major in -> bf16 PIXEL-MAJOR out ([B][1024][256]).
// One block per (batch, group): 8 channels x 1024 pixels.
// ---------------------------------------------------------------------------
__global__ void gn_silu_kernel(const float* __restrict__ in, const float* __restrict__ film,
                               const float* __restrict__ gamma, const float* __restrict__ beta,
                               bf16_t* __restrict__ outb, bf16_t* __restrict__ xcopy,
                               int do_silu) {
  __shared__ float r1[256], r2[256];
  int tid = threadIdx.x;
  int b = blockIdx.x >> 5;
  int g = blockIdx.x & 31;
  int c0 = g * 8;
  size_t base = ((size_t)b * 256 + c0) * 1024;
  float s = 0.f, s2 = 0.f;
  for (int i = tid; i < 8192; i += 256) {
    int c = i >> 10;
    float f = film ? film[b * 256 + c0 + c] : 0.f;
    float v = in[base + i] + f;
    s += v; s2 += v * v;
  }
  r1[tid] = s; r2[tid] = s2;
  __syncthreads();
  for (int o = 128; o > 0; o >>= 1) {
    if (tid < o) { r1[tid] += r1[tid + o]; r2[tid] += r2[tid + o]; }
    __syncthreads();
  }
  float mean = r1[0] * (1.f / 8192.f);
  float var  = r2[0] * (1.f / 8192.f) - mean * mean;
  float rstd = rsqrtf(var + 1e-5f);
  for (int i = tid; i < 8192; i += 256) {
    int c = i >> 10;
    int pix = i & 1023;
    float f = film ? film[b * 256 + c0 + c] : 0.f;
    float v = in[base + i] + f;
    float y = (v - mean) * rstd * gamma[c0 + c] + beta[c0 + c];
    if (do_silu) y = y * (1.f / (1.f + __expf(-y)));
    size_t o = ((size_t)b * 1024 + pix) * 256 + c0 + c;   // pixel-major
    outb[o] = (bf16_t)y;
    if (xcopy) xcopy[o] = (bf16_t)in[base + i];
  }
}

// ---------------------------------------------------------------------------
// 3x3 conv as 9 shifted 1x1 GEMMs over pixel-major bf16 activations.
// out[b,co,y,x] (f32 channel-major) = sum_p sum_ci Wp[p,co,ci]*act[b,(y,x)+shift,ci]
// Block tile 128(Co) x 128(N); 8 waves, each wave 64x32 (4x2 WMMA accums).
// Double-buffered async global->LDS staging; OOB taps read a zeroed pad buffer
// (branchless address select keeps ASYNCcnt uniform: exactly 4 copies/step).
// ---------------------------------------------------------------------------
__global__ void conv3x3_kernel(const bf16_t* __restrict__ act, const bf16_t* __restrict__ Wp,
                               const float* __restrict__ bias, const bf16_t* __restrict__ zpad,
                               float* __restrict__ out) {
  __shared__ __align__(16) bf16_t Al[2][128 * KP];
  __shared__ __align__(16) bf16_t Bl[2][128 * KP];
  int tid = threadIdx.x;
  int lane = tid & 31, wave = tid >> 5;
  int lh = lane >> 4, lm = lane & 15;
  int wm = wave >> 2, wn = wave & 3;
  int ng = blockIdx.x * 128;
  int b = ng >> 10;
  int n0 = ng & 1023;
  int m0 = blockIdx.y * 128;

  v8f acc[4][2] = {};
  int am = tid >> 1, ak = (tid & 1) * 16;    // A-tile fill indices
  int nb = tid >> 1, kh = (tid & 1) * 16;    // B-tile fill indices
  int y = (n0 + nb) >> 5, x = (n0 + nb) & 31;
  const bf16_t* actb = act + (size_t)b * 1024 * 256;
  unsigned ldsa[2] = { lds_addr(&Al[0][am * KP + ak]), lds_addr(&Al[1][am * KP + ak]) };
  unsigned ldsb[2] = { lds_addr(&Bl[0][nb * KP + kh]), lds_addr(&Bl[1][nb * KP + kh]) };

  auto issue = [&](int step, int buf) {
    int p = step >> 3;
    int k0 = (step & 7) * 32;
    const bf16_t* sa = Wp + ((size_t)p * 256 + m0 + am) * 256 + k0 + ak;
    async_b128(ldsa[buf], sa);
    async_b128(ldsa[buf] + 16, sa + 8);
    int dy = p / 3 - 1, dx = p % 3 - 1;
    int yy = y + dy, xx = x + dx;
    bool ok = ((unsigned)yy < 32u) && ((unsigned)xx < 32u);
    const bf16_t* sb = ok ? (actb + (long)(yy * 32 + xx) * 256 + k0 + kh) : zpad;
    async_b128(ldsb[buf], sb);
    async_b128(ldsb[buf] + 16, sb + 8);
  };

  issue(0, 0);
  for (int step = 0; step < 72; step++) {
    int cur = step & 1;
    if (step + 1 < 72) { issue(step + 1, cur ^ 1); wait_async4(); }
    else               { wait_async0(); }
    __syncthreads();
    v16bf af[4], bfr[2];
#pragma unroll
    for (int i = 0; i < 4; i++)
      af[i] = load_frag(&Al[cur][(wm * 64 + i * 16 + lm) * KP], lh);
#pragma unroll
    for (int j = 0; j < 2; j++)
      bfr[j] = load_frag(&Bl[cur][(wn * 32 + j * 16 + lm) * KP], lh);
#pragma unroll
    for (int i = 0; i < 4; i++)
#pragma unroll
      for (int j = 0; j < 2; j++)
        acc[i][j] = wmma_bf16(af[i], bfr[j], acc[i][j]);
    __syncthreads();   // protects buf[cur] before it is refilled two steps later
  }
#pragma unroll
  for (int i = 0; i < 4; i++)
#pragma unroll
    for (int j = 0; j < 2; j++)
#pragma unroll
      for (int e = 0; e < 8; e++) {
        int m = m0 + wm * 64 + i * 16 + lh * 8 + e;
        int n = wn * 32 + j * 16 + lm;
        out[((size_t)b * 256 + m) * 1024 + n0 + n] = acc[i][j][e] + bias[m];
      }
}

// ---------------------------------------------------------------------------
// 1x1 conv GEMM over pixel-major bf16 input: out = A @ Bsrc^T (+bias) (+addend)
//   f32 out  : channel-major [B][M][1024], addend same layout.
//   bf16 out : pixel-major   [B][1024][M]  (feeds later GEMMs / attention).
// M = gridDim.y*128 (256 or 768). Double-buffered async global->LDS staging.
// ---------------------------------------------------------------------------
__global__ void gemm1x1_kernel(const bf16_t* __restrict__ A, const bf16_t* __restrict__ Bsrc,
                               const float* __restrict__ bias, const float* __restrict__ addend,
                               void* __restrict__ out, int out_bf16) {
  __shared__ __align__(16) bf16_t Al[2][128 * KP];
  __shared__ __align__(16) bf16_t Bl[2][128 * KP];
  int tid = threadIdx.x;
  int lane = tid & 31, wave = tid >> 5;
  int lh = lane >> 4, lm = lane & 15;
  int wm = wave >> 2, wn = wave & 3;
  int b = blockIdx.x >> 3;
  int n0 = (blockIdx.x & 7) * 128;
  int m0 = blockIdx.y * 128;
  int Mtot = (int)gridDim.y * 128;

  v8f acc[4][2] = {};
  int am = tid >> 1, ak = (tid & 1) * 16;
  int nb = tid >> 1, kh = (tid & 1) * 16;
  unsigned ldsa[2] = { lds_addr(&Al[0][am * KP + ak]), lds_addr(&Al[1][am * KP + ak]) };
  unsigned ldsb[2] = { lds_addr(&Bl[0][nb * KP + kh]), lds_addr(&Bl[1][nb * KP + kh]) };
  const bf16_t* arow = A + (size_t)(m0 + am) * 256 + ak;
  const bf16_t* brow = Bsrc + ((size_t)b * 1024 + n0 + nb) * 256 + kh;

  auto issue = [&](int step, int buf) {
    int k0 = step * 32;
    async_b128(ldsa[buf], arow + k0);
    async_b128(ldsa[buf] + 16, arow + k0 + 8);
    async_b128(ldsb[buf], brow + k0);
    async_b128(ldsb[buf] + 16, brow + k0 + 8);
  };

  issue(0, 0);
  for (int step = 0; step < 8; step++) {
    int cur = step & 1;
    if (step + 1 < 8) { issue(step + 1, cur ^ 1); wait_async4(); }
    else              { wait_async0(); }
    __syncthreads();
    v16bf af[4], bfr[2];
#pragma unroll
    for (int i = 0; i < 4; i++)
      af[i] = load_frag(&Al[cur][(wm * 64 + i * 16 + lm) * KP], lh);
#pragma unroll
    for (int j = 0; j < 2; j++)
      bfr[j] = load_frag(&Bl[cur][(wn * 32 + j * 16 + lm) * KP], lh);
#pragma unroll
    for (int i = 0; i < 4; i++)
#pragma unroll
      for (int j = 0; j < 2; j++)
        acc[i][j] = wmma_bf16(af[i], bfr[j], acc[i][j]);
    __syncthreads();
  }
#pragma unroll
  for (int i = 0; i < 4; i++)
#pragma unroll
    for (int j = 0; j < 2; j++)
#pragma unroll
      for (int e = 0; e < 8; e++) {
        int m = m0 + wm * 64 + i * 16 + lh * 8 + e;
        int n = wn * 32 + j * 16 + lm;
        float val = acc[i][j][e];
        if (bias) val += bias[m];
        if (out_bf16) {
          ((bf16_t*)out)[((size_t)b * 1024 + n0 + n) * Mtot + m] = (bf16_t)val;
        } else {
          size_t oi = ((size_t)b * Mtot + m) * 1024 + n0 + n;
          if (addend) val += addend[oi];
          ((float*)out)[oi] = val;
        }
      }
}

// ---------------------------------------------------------------------------
// Flash attention, one block per (batch, 64-query tile). d=256, seq=1024.
//   S = (Q^T K)/16  -> streaming softmax -> O += P V^T, final O /= l.
// qkv: bf16 pixel-major [B][1024][768] (q|k|v per pixel).
// out: bf16 pixel-major [B][1024][256].
// ---------------------------------------------------------------------------
__global__ void flash_attn_kernel(const bf16_t* __restrict__ qkv, bf16_t* __restrict__ outp) {
  extern __shared__ __align__(16) char smem[];
  bf16_t* Ql = (bf16_t*)smem;            // [64][QP]  Ql[s][d]
  bf16_t* Kl = Ql + 64 * QP;             // [64][QP]  Kl[t][d]
  bf16_t* Vl = Kl + 64 * QP;             // [256][VP] Vl[d][t]
  bf16_t* Pl = Vl + 256 * VP;            // [64][VP]  Pl[s][t]
  float*  Sl = (float*)(Pl + 64 * VP);   // [64][64]
  float*  mrun = Sl + 64 * 64;           // [64]
  float*  lrun = mrun + 64;              // [64]
  float*  scl  = lrun + 64;              // [64]
  float*  red  = scl + 64;               // [64][4]

  int tid = threadIdx.x;
  int lane = tid & 31, wave = tid >> 5;
  int lh = lane >> 4, lm = lane & 15;
  int b = blockIdx.y;
  int s0 = blockIdx.x * 64;
  const bf16_t* base = qkv + (size_t)b * 1024 * 768;

  { // Q tile: Ql[s][d] = qkv[b][s0+s][d], contiguous vector loads
    int s = tid >> 2, d0 = (tid & 3) * 64;
    const bf16_t* src = base + (size_t)(s0 + s) * 768 + d0;
#pragma unroll
    for (int i = 0; i < 64; i += 8)
      *(bf16x8*)&Ql[s * QP + d0 + i] = *(const bf16x8*)(src + i);
  }
  if (tid < 64) { mrun[tid] = -1e30f; lrun[tid] = 0.f; }
  __syncthreads();

  int mt = wave >> 1;     // query 16-row tile (0..3)
  int half = wave & 1;    // d-half for O / t-half pair for S

  v8f acc[8] = {};        // O accum: M=mt tile, N = half*128 + j*16

  for (int t0 = 0; t0 < 1024; t0 += 64) {
    { // K tile: Kl[t][d], contiguous
      int t = tid >> 2, d0 = (tid & 3) * 64;
      const bf16_t* src = base + (size_t)(t0 + t) * 768 + 256 + d0;
#pragma unroll
      for (int i = 0; i < 64; i += 8)
        *(bf16x8*)&Kl[t * QP + d0 + i] = *(const bf16x8*)(src + i);
    }
    { // V tile transposed into Vl[d][t]: vector-load 8 channels, scatter to LDS
#pragma unroll
      for (int r = 0; r < 8; r++) {
        int task = tid + r * 256;
        int t = task >> 5;            // 0..63
        int d0 = (task & 31) * 8;     // 0..248
        bf16x8 v = *(const bf16x8*)(base + (size_t)(t0 + t) * 768 + 512 + d0);
#pragma unroll
        for (int u = 0; u < 8; u++) Vl[(d0 + u) * VP + t] = v[u];
      }
    }
    __syncthreads();

    // S tile (64x64): wave does tiles (mt, half*2 + {0,1}) over K=256
    v8f sacc[2] = {};
#pragma unroll
    for (int k0 = 0; k0 < 256; k0 += 32) {
      v16bf af = load_frag(&Ql[(mt * 16 + lm) * QP + k0], lh);
#pragma unroll
      for (int j = 0; j < 2; j++) {
        v16bf bfr = load_frag(&Kl[((half * 2 + j) * 16 + lm) * QP + k0], lh);
        sacc[j] = wmma_bf16(af, bfr, sacc[j]);
      }
    }
#pragma unroll
    for (int j = 0; j < 2; j++)
#pragma unroll
      for (int e = 0; e < 8; e++)
        Sl[(mt * 16 + lh * 8 + e) * 64 + (half * 2 + j) * 16 + lm] =
            sacc[j][e] * 0.0625f;   // 1/sqrt(256)
    __syncthreads();

    // streaming softmax
    { int r = tid >> 2, seg = tid & 3;
      float pm = -1e30f;
      for (int c = seg * 16; c < seg * 16 + 16; c++) pm = fmaxf(pm, Sl[r * 64 + c]);
      red[r * 4 + seg] = pm; }
    __syncthreads();
    if (tid < 64) {
      int r = tid;
      float tm = fmaxf(fmaxf(red[r * 4], red[r * 4 + 1]),
                       fmaxf(red[r * 4 + 2], red[r * 4 + 3]));
      float mnew = fmaxf(mrun[r], tm);
      scl[r] = __expf(mrun[r] - mnew);
      mrun[r] = mnew;
    }
    __syncthreads();
    { int r = tid >> 2, seg = tid & 3;
      float m = mrun[r], ps = 0.f;
      for (int c = seg * 16; c < seg * 16 + 16; c++) {
        float p = __expf(Sl[r * 64 + c] - m);
        Pl[r * VP + c] = (bf16_t)p;
        ps += p;
      }
      red[r * 4 + seg] = ps; }
    __syncthreads();
    if (tid < 64) {
      int r = tid;
      lrun[r] = lrun[r] * scl[r] + red[r * 4] + red[r * 4 + 1] + red[r * 4 + 2] + red[r * 4 + 3];
    }
    __syncthreads();

    // rescale O accum, then O += P * V^T (K = 64)
    float sc[8];
#pragma unroll
    for (int e = 0; e < 8; e++) sc[e] = scl[mt * 16 + lh * 8 + e];
#pragma unroll
    for (int j = 0; j < 8; j++)
#pragma unroll
      for (int e = 0; e < 8; e++) acc[j][e] *= sc[e];
#pragma unroll
    for (int k0 = 0; k0 < 64; k0 += 32) {
      v16bf pf = load_frag(&Pl[(mt * 16 + lm) * VP + k0], lh);
#pragma unroll
      for (int j = 0; j < 8; j++) {
        v16bf vf = load_frag(&Vl[(half * 128 + j * 16 + lm) * VP + k0], lh);
        acc[j] = wmma_bf16(pf, vf, acc[j]);
      }
    }
    __syncthreads();
  }

  // epilogue: O /= l ; store pixel-major [B][s][d] bf16
  float rl[8];
#pragma unroll
  for (int e = 0; e < 8; e++) rl[e] = 1.f / lrun[mt * 16 + lh * 8 + e];
  bf16_t* ob = outp + (size_t)b * 1024 * 256;
#pragma unroll
  for (int j = 0; j < 8; j++) {
    int d = half * 128 + j * 16 + lm;
#pragma unroll
    for (int e = 0; e < 8; e++) {
      int s = mt * 16 + lh * 8 + e;
      ob[(size_t)(s0 + s) * 256 + d] = (bf16_t)(acc[j][e] * rl[e]);
    }
  }
}

// ---------------------------------------------------------------------------
// Host-side orchestration
// ---------------------------------------------------------------------------
extern "C" void kernel_launch(void* const* d_in, const int* in_sizes, int n_in,
                              void* d_out, int out_size, void* d_ws, size_t ws_size,
                              hipStream_t stream) {
  (void)in_sizes; (void)n_in; (void)out_size; (void)ws_size;
  const float* x      = (const float*)d_in[0];
  const float* ne     = (const float*)d_in[1];
  const float* gn1_s  = (const float*)d_in[2];
  const float* gn1_b  = (const float*)d_in[3];
  const float* conv1w = (const float*)d_in[4];
  const float* conv1b = (const float*)d_in[5];
  const float* noisew = (const float*)d_in[6];
  const float* noiseb = (const float*)d_in[7];
  const float* gn2_s  = (const float*)d_in[8];
  const float* gn2_b  = (const float*)d_in[9];
  const float* conv2w = (const float*)d_in[10];
  const float* conv2b = (const float*)d_in[11];
  const float* resw   = (const float*)d_in[12];
  const float* resb   = (const float*)d_in[13];
  const float* agn_s  = (const float*)d_in[14];
  const float* agn_b  = (const float*)d_in[15];
  const float* qkvw   = (const float*)d_in[16];
  const float* outw   = (const float*)d_in[17];
  const float* outb   = (const float*)d_in[18];
  float* out = (float*)d_out;

  char* ws = (char*)d_ws;
  bf16_t* a1    = (bf16_t*)(ws + 0);          // bf16 pm [16][1024][256], reused as attnout
  bf16_t* xb    = (bf16_t*)(ws + 8388608);    // bf16 pm copy of x
  float*  h1    = (float*) (ws + 16777216);   // f32 cm, conv1 out, later h (post-res)
  bf16_t* a2    = (bf16_t*)(ws + 33554432);   // bf16 pm, gn2 out, reused as a3
  float*  h2    = (float*) (ws + 41943040);   // f32 cm, conv2 out
  bf16_t* qkv   = (bf16_t*)(ws + 58720256);   // bf16 pm [16][1024][768]
  float*  film  = (float*) (ws + 83886080);   // [16][256]
  bf16_t* w1p   = (bf16_t*)(ws + 83902464);   // [9][256][256]
  bf16_t* w2p   = (bf16_t*)(ws + 85082112);
  bf16_t* reswp = (bf16_t*)(ws + 86261760);   // [256][256]
  bf16_t* qkvwp = (bf16_t*)(ws + 86392832);   // [768][256]
  bf16_t* outwp = (bf16_t*)(ws + 86786048);   // [256][256]
  bf16_t* zpad  = (bf16_t*)(ws + 86917120);   // 256B zeroed pad (OOB redirect)
  bf16_t* attnout = a1;
  bf16_t* a3 = a2;

  // weight repack + zero pad
  pack3x3_kernel<<<2304, 256, 0, stream>>>(conv1w, w1p);
  pack3x3_kernel<<<2304, 256, 0, stream>>>(conv2w, w2p);
  pack_cast_kernel<<<256, 256, 0, stream>>>(resw, reswp, 65536);
  pack_cast_kernel<<<768, 256, 0, stream>>>(qkvw, qkvwp, 196608);
  pack_cast_kernel<<<256, 256, 0, stream>>>(outw, outwp, 65536);
  zero_pad_kernel<<<1, 64, 0, stream>>>((float*)zpad);
  film_kernel<<<16, 256, 0, stream>>>(ne, noisew, noiseb, film);

  // ResnetBlock
  gn_silu_kernel<<<512, 256, 0, stream>>>(x, nullptr, gn1_s, gn1_b, a1, xb, 1);
  conv3x3_kernel<<<dim3(128, 2), 256, 0, stream>>>(a1, w1p, conv1b, zpad, h1);
  gn_silu_kernel<<<512, 256, 0, stream>>>(h1, film, gn2_s, gn2_b, a2, nullptr, 1);
  conv3x3_kernel<<<dim3(128, 2), 256, 0, stream>>>(a2, w2p, conv2b, zpad, h2);
  // h = conv2 out + 1x1 residual conv(x)
  gemm1x1_kernel<<<dim3(128, 2), 256, 0, stream>>>(reswp, xb, resb, h2, h1, 0);

  // SelfAttention
  gn_silu_kernel<<<512, 256, 0, stream>>>(h1, nullptr, agn_s, agn_b, a3, nullptr, 0);
  gemm1x1_kernel<<<dim3(128, 6), 256, 0, stream>>>(qkvwp, a3, nullptr, nullptr, qkv, 1);
  size_t fl_smem = (size_t)64 * QP * 2 * 2 + (size_t)256 * VP * 2 + (size_t)64 * VP * 2 +
                   64 * 64 * 4 + (64 * 3 + 256) * 4;   // 131840 B
  flash_attn_kernel<<<dim3(16, 16), 256, fl_smem, stream>>>(qkv, attnout);
  // out = out_w @ attnout + out_b + h
  gemm1x1_kernel<<<dim3(128, 2), 256, 0, stream>>>(outwp, attnout, outb, h1, out, 0);
}
